// AttentionLayer_1700807049606
// MI455X (gfx1250) — matrix operand order repaired and checked by reading, not compiled, not run on hardware
//
#include <hip/hip_runtime.h>
#include <hip/hip_bf16.h>

#define BB 32
#define NN 2048
#define FF 128
#define DD 64

typedef _Float16 v16h __attribute__((ext_vector_type(16)));
typedef float    v8f  __attribute__((ext_vector_type(8)));

__device__ __forceinline__ v8f zero8() {
    v8f z;
    #pragma unroll
    for (int i = 0; i < 8; ++i) z[i] = 0.0f;
    return z;
}

// A operand (16x32, f16) from row-major [row][k] storage, leading dim ld (halfs).
__device__ __forceinline__ v16h load_a_f16(const _Float16* base, int ld, int row0,
                                           int k0, int l16, int half) {
    const _Float16* p0 = base + (size_t)(row0 + l16) * ld + k0 + half * 8;
    v16h a;
    *(uint4*)&a       = *(const uint4*)p0;
    *((uint4*)&a + 1) = *(const uint4*)(p0 + 16);
    return a;
}

// Same A layout but source is f32 row-major (convert to f16).
__device__ __forceinline__ v16h load_a_f32(const float* base, int ld, int row0,
                                           int k0, int l16, int half) {
    const float* p0 = base + (size_t)(row0 + l16) * ld + k0 + half * 8;
    float4 f0 = *(const float4*)(p0);
    float4 f1 = *(const float4*)(p0 + 4);
    float4 f2 = *(const float4*)(p0 + 16);
    float4 f3 = *(const float4*)(p0 + 20);
    v16h a;
    a[0]=(_Float16)f0.x; a[1]=(_Float16)f0.y; a[2]=(_Float16)f0.z; a[3]=(_Float16)f0.w;
    a[4]=(_Float16)f1.x; a[5]=(_Float16)f1.y; a[6]=(_Float16)f1.z; a[7]=(_Float16)f1.w;
    a[8]=(_Float16)f2.x; a[9]=(_Float16)f2.y; a[10]=(_Float16)f2.z; a[11]=(_Float16)f2.w;
    a[12]=(_Float16)f3.x; a[13]=(_Float16)f3.y; a[14]=(_Float16)f3.z; a[15]=(_Float16)f3.w;
    return a;
}

// B operand (32x16, f16) where source holds B-transposed row-major: src[n][k].
__device__ __forceinline__ v16h load_bt(const _Float16* base, int ld, int n0,
                                        int k0, int l16, int half) {
    const _Float16* p = base + (size_t)(n0 + l16) * ld + k0 + half * 8;
    v16h b;
    *(uint4*)&b       = *(const uint4*)p;
    *((uint4*)&b + 1) = *(const uint4*)(p + 16);
    return b;
}

__device__ __forceinline__ v8f wmma16(v16h a, v16h b, v8f c) {
    return __builtin_amdgcn_wmma_f32_16x16x32_f16(false, a, false, b,
                                                  (short)0, c, false, false);
}

// Async 16B-per-lane copy: global -> LDS, tracked by ASYNCcnt (CDNA5 TDM-lite path).
__device__ __forceinline__ void async_copy16(unsigned int lds_off, const void* gptr) {
    asm volatile("global_load_async_to_lds_b128 %0, %1, off"
                 :: "v"(lds_off), "v"((unsigned long long)(uintptr_t)gptr)
                 : "memory");
}
__device__ __forceinline__ void wait_async0() {
    asm volatile("s_wait_asynccnt 0x0" ::: "memory");
}

// ---------------- Kernel 0: pack adjacency into bitmask ----------------
__global__ __launch_bounds__(256) void pack_adj_kernel(const int* __restrict__ adj,
                                                       unsigned int* __restrict__ bits) {
    int w = blockIdx.x * 256 + threadIdx.x;          // word index < 2048*64
    const int* p = adj + (size_t)w * 32;             // 32 consecutive ints (128B)
    unsigned int m = 0;
    #pragma unroll
    for (int i = 0; i < 32; ++i)
        m |= (p[i] != 0 ? 1u : 0u) << i;
    bits[w] = m;
}

// ---------------- Kernel 1: QKV projections ----------------
__global__ __launch_bounds__(256) void qkv_kernel(const float* __restrict__ x,
                                                  const float* __restrict__ WQ,
                                                  const float* __restrict__ WK,
                                                  const float* __restrict__ WV,
                                                  _Float16* __restrict__ Qh,
                                                  _Float16* __restrict__ Kh,
                                                  _Float16* __restrict__ Vt) {
    __shared__ _Float16 wt[3 * DD * FF];  // W transposed: [mat][d][f], f16
    const int tid = threadIdx.x;
    const float* Ws[3] = {WQ, WK, WV};
    #pragma unroll
    for (int m = 0; m < 3; ++m) {
        const float* W = Ws[m];
        for (int idx = tid; idx < FF * DD; idx += 256) {
            int f = idx >> 6, d = idx & 63;
            wt[m * (DD * FF) + d * FF + f] = (_Float16)W[idx];
        }
    }
    __syncthreads();

    const int wave = tid >> 5, lane = tid & 31;
    const int l16 = lane & 15, half = lane >> 4;
    const int r0 = (blockIdx.x * 8 + wave) * 16;  // global row in [0, B*N)
    const int b  = r0 >> 11;
    const int n0 = r0 & (NN - 1);

    v16h a[4];
    #pragma unroll
    for (int kc = 0; kc < 4; ++kc) a[kc] = load_a_f32(x, FF, r0, kc * 32, l16, half);

    #pragma unroll
    for (int m = 0; m < 3; ++m) {
        const _Float16* wtm = &wt[m * (DD * FF)];
        #pragma unroll
        for (int t = 0; t < 4; ++t) {
            v8f c = zero8();
            #pragma unroll
            for (int kc = 0; kc < 4; ++kc) {
                v16h bf = load_bt(wtm, FF, t * 16, kc * 32, l16, half);
                c = wmma16(a[kc], bf, c);
            }
            if (m < 2) {
                _Float16* dst = (m == 0 ? Qh : Kh) + (size_t)r0 * DD;
                #pragma unroll
                for (int v = 0; v < 8; ++v) {
                    int row = v + 8 * half;
                    dst[row * DD + t * 16 + l16] = (_Float16)c[v];
                }
            } else {
                _Float16* dst = Vt + (size_t)b * DD * NN;  // Vt[b][d][n]
                #pragma unroll
                for (int v = 0; v < 8; ++v) {
                    int row = v + 8 * half;
                    dst[(size_t)(t * 16 + l16) * NN + (n0 + row)] = (_Float16)c[v];
                }
            }
        }
    }
}

// ---------------- Kernel 2: fused masked attention ----------------
// Double-buffered K/V tiles filled by async global->LDS copies; compute overlaps
// the next tile's DMA; one barrier + one s_wait_asynccnt per iteration.
__global__ __launch_bounds__(128) void attn_kernel(const _Float16* __restrict__ Qh,
                                                   const _Float16* __restrict__ Kh,
                                                   const _Float16* __restrict__ Vtg,
                                                   const unsigned int* __restrict__ adjb,
                                                   float* __restrict__ out) {
    __shared__ _Float16 ktile[2][64 * 64];   // K tile  [m_local][d]
    __shared__ _Float16 vtile[2][64 * 64];   // Vt tile [d][m_local]
    __shared__ _Float16 plds[4 * 16 * 64];   // per-wave P staging (16x64 row-major)

    const int tid = threadIdx.x, wave = tid >> 5, lane = tid & 31;
    const int l16 = lane & 15, half = lane >> 4;
    const int b  = blockIdx.x >> 5;
    const int rb = blockIdx.x & 31;
    const int nb = rb * 64 + wave * 16;

    const _Float16* Qb = Qh  + (size_t)b * NN * DD;
    const _Float16* Kb = Kh  + (size_t)b * NN * DD;
    const _Float16* Vb = Vtg + (size_t)b * DD * NN;
    _Float16* pw = &plds[wave * (16 * 64)];

    // Stage tile (mbase) into buffer buf via async copies. Per wave: 4 K + 4 V chunks.
    auto stage_async = [&](int mbase, int buf) {
        const char* kg = (const char*)(Kb + (size_t)mbase * DD);  // contiguous 8KB
        #pragma unroll
        for (int j = 0; j < 4; ++j) {
            int byteoff = (wave + 4 * j) * 512 + lane * 16;
            unsigned int klds =
                (unsigned int)(uintptr_t)((const char*)&ktile[buf][0] + byteoff);
            async_copy16(klds, kg + byteoff);
        }
        #pragma unroll
        for (int j = 0; j < 4; ++j) {
            int byteoff = (wave + 4 * j) * 512 + lane * 16;
            int d = byteoff >> 7, col = byteoff & 127;        // vtile row / byte col
            unsigned int vlds =
                (unsigned int)(uintptr_t)((const char*)&vtile[buf][0] + byteoff);
            async_copy16(vlds,
                         (const char*)Vb + (size_t)d * (NN * 2) + mbase * 2 + col);
        }
    };

    v16h aQ[2];
    aQ[0] = load_a_f16(Qb, DD, nb, 0, l16, half);
    aQ[1] = load_a_f16(Qb, DD, nb, 32, l16, half);

    float mrow[8], lrow[8];
    v8f acc[4];
    #pragma unroll
    for (int v = 0; v < 8; ++v) { mrow[v] = -1e20f; lrow[v] = 0.0f; }
    #pragma unroll
    for (int t = 0; t < 4; ++t) acc[t] = zero8();

    const float scale = 0.125f;  // 1/sqrt(64)

    // Prologue: fill buffer 0.
    stage_async(0, 0);
    wait_async0();
    __syncthreads();

    for (int mt = 0; mt < NN / 64; ++mt) {
        const int buf = mt & 1;

        // Kick off DMA for the next tile into the other buffer (overlaps compute).
        if (mt + 1 < NN / 64) stage_async((mt + 1) * 64, buf ^ 1);

        // ---- S = Q K^T tile (16 x 64), f32 accumulate ----
        v8f s[4];
        #pragma unroll
        for (int nt = 0; nt < 4; ++nt) {
            v8f c = zero8();
            #pragma unroll
            for (int kc = 0; kc < 2; ++kc) {
                v16h bK = load_bt(&ktile[buf][0], 64, nt * 16, kc * 32, l16, half);
                c = wmma16(aQ[kc], bK, c);
            }
            s[nt] = c;
        }

        // ---- scale + adjacency mask from packed bits ----
        #pragma unroll
        for (int v = 0; v < 8; ++v) {
            int row = nb + v + 8 * half;
            uint2 wv = *(const uint2*)(adjb + (size_t)row * (NN / 32) + mt * 2);
            #pragma unroll
            for (int nt = 0; nt < 4; ++nt) {
                unsigned int w = (nt < 2) ? wv.x : wv.y;
                int bit = ((nt & 1) << 4) + l16;
                float e = s[nt][v] * scale;
                s[nt][v] = ((w >> bit) & 1u) ? e : -1e20f;
            }
        }

        // ---- online softmax; each row lives across one 16-lane half ----
        #pragma unroll
        for (int v = 0; v < 8; ++v) {
            float tmax = fmaxf(fmaxf(s[0][v], s[1][v]), fmaxf(s[2][v], s[3][v]));
            #pragma unroll
            for (int off = 1; off < 16; off <<= 1)
                tmax = fmaxf(tmax, __shfl_xor(tmax, off, 32));
            float newm = fmaxf(mrow[v], tmax);
            float fsc  = __expf(mrow[v] - newm);
            mrow[v] = newm;
            float psum = 0.0f;
            #pragma unroll
            for (int nt = 0; nt < 4; ++nt) {
                float p = __expf(s[nt][v] - newm);
                s[nt][v] = p;
                psum += p;
            }
            #pragma unroll
            for (int off = 1; off < 16; off <<= 1)
                psum += __shfl_xor(psum, off, 32);
            lrow[v] = lrow[v] * fsc + psum;
            #pragma unroll
            for (int nt = 0; nt < 4; ++nt) acc[nt][v] *= fsc;
        }

        // ---- stage P (C-layout f32 -> row-major f16 in LDS), re-read in A-layout ----
        #pragma unroll
        for (int nt = 0; nt < 4; ++nt)
            #pragma unroll
            for (int v = 0; v < 8; ++v)
                pw[(v + 8 * half) * 64 + nt * 16 + l16] = (_Float16)s[nt][v];
        asm volatile("s_wait_dscnt 0" ::: "memory");

        v16h aP0 = load_a_f16(pw, 64, 0, 0, l16, half);
        v16h aP1 = load_a_f16(pw, 64, 0, 32, l16, half);

        // ---- O += P @ V  (B operand from LDS Vt tile) ----
        #pragma unroll
        for (int nt = 0; nt < 4; ++nt) {
            v8f c = acc[nt];
            v16h b0 = load_bt(&vtile[buf][0], 64, nt * 16, 0, l16, half);
            c = wmma16(aP0, b0, c);
            v16h b1 = load_bt(&vtile[buf][0], 64, nt * 16, 32, l16, half);
            c = wmma16(aP1, b1, c);
            acc[nt] = c;
        }

        // Next tile's DMA must have landed before anyone reads it.
        wait_async0();
        __syncthreads();
    }

    // ---- finalize: H = O / l ----
    float* ob = out + (size_t)b * NN * DD;
    #pragma unroll
    for (int v = 0; v < 8; ++v) {
        float inv = 1.0f / lrow[v];
        int row = nb + v + 8 * half;
        #pragma unroll
        for (int nt = 0; nt < 4; ++nt)
            ob[(size_t)row * DD + nt * 16 + l16] = acc[nt][v] * inv;
    }
}

extern "C" void kernel_launch(void* const* d_in, const int* in_sizes, int n_in,
                              void* d_out, int out_size, void* d_ws, size_t ws_size,
                              hipStream_t stream) {
    (void)in_sizes; (void)n_in; (void)out_size; (void)ws_size;
    const float* x   = (const float*)d_in[0];
    const int*   adj = (const int*)d_in[1];
    const float* WQ  = (const float*)d_in[2];
    const float* WK  = (const float*)d_in[3];
    const float* WV  = (const float*)d_in[4];
    float* out = (float*)d_out;

    const size_t elems = (size_t)BB * NN * DD;  // 4M halfs each
    _Float16* Qh = (_Float16*)d_ws;
    _Float16* Kh = Qh + elems;
    _Float16* Vt = Kh + elems;
    unsigned int* adjb = (unsigned int*)(Vt + elems);  // +512 KB; total ws ~24.5 MB

    pack_adj_kernel<<<(NN * NN / 32) / 256, 256, 0, stream>>>(adj, adjb);
    qkv_kernel<<<(BB * NN) / (16 * 8), 256, 0, stream>>>(x, WQ, WK, WV, Qh, Kh, Vt);
    attn_kernel<<<BB * (NN / 64), 128, 0, stream>>>(Qh, Kh, Vt, adjb, out);
}
